// WindowAttention_53102975648177
// MI455X (gfx1250) — matrix operand order repaired and checked by reading, not compiled, not run on hardware
//
#include <hip/hip_runtime.h>

// ---------------------------------------------------------------------------
// Fused Swin window attention for MI455X (gfx1250, wave32, WMMA).
//   kernel 1: convert W{q,k,v,o} (fp32, KxN) -> W^T (bf16, NxK) in d_ws
//   kernel 2: one block per 7x7 window, everything (QKV, attention, out-proj)
//             fused through LDS using v_wmma_f32_16x16x32_bf16.
// Round 2 change: use hardware bf16 converts (v_cvt_pk_bf16_f32) instead of
// manual RNE bit-twiddling to cut VALU in the GEMM epilogues.
// ---------------------------------------------------------------------------

#define DIMC   192
#define NHEAD  6
#define HD     32
#define LTOK   49
#define LPAD   64
#define WSZ    7
#define IMG    56
#define ATT_SCALE 0.17677669529663687f  // 32^-0.5

#define SX_STR  200   // bf16 row stride for 192-wide token arrays (bank pad)
#define SVT_STR 72    // bf16 row stride for 64-wide token arrays
#define SP_STR  72
#define NWAVE   8

typedef __attribute__((ext_vector_type(16))) __bf16 v16bf;
typedef __attribute__((ext_vector_type(8)))  float  v8f;
typedef __attribute__((ext_vector_type(2)))  float  v2f;
typedef __attribute__((ext_vector_type(2)))  __bf16 v2bf;

union BFTile { v16bf v; uint4 q[2]; };

__device__ __forceinline__ unsigned short f2bf(float f) {
  __bf16 h = static_cast<__bf16>(f);       // v_cvt bf16 (RNE)
  unsigned short u;
  __builtin_memcpy(&u, &h, 2);
  return u;
}
__device__ __forceinline__ unsigned pack2(float a, float b) {
  v2f v = {a, b};
  v2bf h = __builtin_convertvector(v, v2bf);  // v_cvt_pk_bf16_f32
  unsigned u;
  __builtin_memcpy(&u, &h, 4);
  return u;
}

// A operand (16x32), row-major MxK source. Lane r<16: K = 8h..8h+7 then 16+8h..
__device__ __forceinline__ v16bf load_tile_a(const unsigned short* p0, int stride, int lane) {
  const int half = lane >> 4, lr = lane & 15;
  const unsigned short* p = p0 + lr * stride + 8 * half;
  BFTile t;
  t.q[0] = *(const uint4*)(p);
  t.q[1] = *(const uint4*)(p + 16);
  return t.v;
}
// B operand (32x16), source is row-major NxK (pre-transposed). Lane n<16: K = 8h..8h+7
// then 16+8h.. (same order as A).
__device__ __forceinline__ v16bf load_tile_b(const unsigned short* p0, int stride, int lane) {
  const int half = lane >> 4, lr = lane & 15;
  const unsigned short* p = p0 + lr * stride + 8 * half;
  BFTile t;
  t.q[0] = *(const uint4*)(p);
  t.q[1] = *(const uint4*)(p + 16);
  return t.v;
}

#define WMMA_BF16(A, B, C) \
  __builtin_amdgcn_wmma_f32_16x16x32_bf16(false, (A), false, (B), (short)0, (C), false, false)

// ---------------------------------------------------------------------------
__global__ void prep_weights_kernel(const float* __restrict__ Wq, const float* __restrict__ Wk,
                                    const float* __restrict__ Wv, const float* __restrict__ Wo,
                                    unsigned short* __restrict__ wt)
{
  int tid = blockIdx.x * blockDim.x + threadIdx.x;
  if (tid >= 4 * DIMC * DIMC) return;
  int m   = tid / (DIMC * DIMC);
  int rem = tid - m * (DIMC * DIMC);
  int n   = rem / DIMC;          // output channel (row of W^T)
  int k   = rem - n * DIMC;      // input channel  (col of W^T, contiguous)
  const float* W = (m == 0) ? Wq : (m == 1) ? Wk : (m == 2) ? Wv : Wo;
  wt[tid] = f2bf(W[k * DIMC + n]);
}

// ---------------------------------------------------------------------------
__global__ __launch_bounds__(256, 1)
void win_attn_kernel(const float* __restrict__ x,
                     const float* __restrict__ bq, const float* __restrict__ bk,
                     const float* __restrict__ bv, const float* __restrict__ bo,
                     const unsigned short* __restrict__ wtq,
                     const unsigned short* __restrict__ wtk,
                     const unsigned short* __restrict__ wtv,
                     const unsigned short* __restrict__ wto,
                     float* __restrict__ out)
{
  extern __shared__ __align__(16) unsigned short smem[];
  unsigned short* sX  = smem;                    // LPAD x SX_STR  (x bf16; later attn out)
  unsigned short* sQ  = sX  + LPAD * SX_STR;     // LPAD x SX_STR
  unsigned short* sK  = sQ  + LPAD * SX_STR;     // LPAD x SX_STR
  unsigned short* sVT = sK  + LPAD * SX_STR;     // DIMC x SVT_STR (V transposed: dim x token)
  unsigned short* sP  = sVT + DIMC * SVT_STR;    // NWAVE x 16 x SP_STR (softmax probs)

  const int tid  = threadIdx.x;
  const int lane = tid & 31;
  const int wave = tid >> 5;
  const int half = lane >> 4;
  const int lr   = lane & 15;

  const int bi    = blockIdx.x;
  const int batch = bi >> 6;
  const int wh    = (bi >> 3) & 7;
  const int ww    = bi & 7;
  const float* xwin = x   + ((size_t)batch * IMG * IMG + (size_t)wh * WSZ * IMG + (size_t)ww * WSZ) * DIMC;
  float*       owin = out + ((size_t)batch * IMG * IMG + (size_t)wh * WSZ * IMG + (size_t)ww * WSZ) * DIMC;

  // ---- stage x window -> LDS bf16, zero pad rows 49..63 --------------------
  for (int e = tid; e < LPAD * (DIMC / 4); e += 256) {
    int token = e / (DIMC / 4);
    int c4    = (e - token * (DIMC / 4)) * 4;
    float4 vv = make_float4(0.f, 0.f, 0.f, 0.f);
    if (token < LTOK) {
      int ty = token / WSZ, tx = token - ty * WSZ;
      vv = *(const float4*)(xwin + (ty * IMG + tx) * DIMC + c4);
    }
    uint2 pk;
    pk.x = pack2(vv.x, vv.y);
    pk.y = pack2(vv.z, vv.w);
    *(uint2*)(sX + token * SX_STR + c4) = pk;
  }
  __syncthreads();

  // ---- Q, K, V projections (M=64 pad, N=192, K=192) ------------------------
  for (int p = 0; p < 3; ++p) {
    const unsigned short* WT = (p == 0) ? wtq : (p == 1) ? wtk : wtv;
    const float* bias        = (p == 0) ? bq  : (p == 1) ? bk  : bv;
    for (int t = wave; t < 48; t += NWAVE) {
      int mt = t / 12, nt = t - mt * 12;
      v8f acc = {};
      #pragma unroll
      for (int ks = 0; ks < 6; ++ks) {
        v16bf a = load_tile_a(sX + mt * 16 * SX_STR + ks * 32, SX_STR, lane);
        v16bf b = load_tile_b(WT + nt * 16 * DIMC  + ks * 32, DIMC,  lane);
        acc = WMMA_BF16(a, b, acc);
      }
      float bb = bias[nt * 16 + lr];
      if (p == 2) {
        // store V transposed: row = out-dim, col = token (8 contiguous tokens/lane)
        unsigned short* d = sVT + (nt * 16 + lr) * SVT_STR + mt * 16 + half * 8;
        uint4 pk;
        pk.x = pack2(acc[0] + bb, acc[1] + bb);
        pk.y = pack2(acc[2] + bb, acc[3] + bb);
        pk.z = pack2(acc[4] + bb, acc[5] + bb);
        pk.w = pack2(acc[6] + bb, acc[7] + bb);
        *(uint4*)d = pk;
      } else {
        unsigned short* sD = (p == 0) ? sQ : sK;
        #pragma unroll
        for (int r = 0; r < 8; ++r)
          sD[(mt * 16 + half * 8 + r) * SX_STR + nt * 16 + lr] = f2bf(acc[r] + bb);
      }
    }
  }
  __syncthreads();

  // ---- attention: 24 tasks = (head h, query M-tile mt) ---------------------
  for (int task = wave; task < 24; task += NWAVE) {
    int h = task >> 2, mt = task & 3;
    v16bf aq = load_tile_a(sQ + mt * 16 * SX_STR + h * HD, SX_STR, lane);

    v8f sc[4];                                    // scores, 16x64 per wave
    #pragma unroll
    for (int nt = 0; nt < 4; ++nt) {
      // B = K^T tile: columns = key tokens (rows of sK), K = head dims (contiguous)
      v16bf bk_ = load_tile_b(sK + nt * 16 * SX_STR + h * HD, SX_STR, lane);
      v8f z = {};
      sc[nt] = WMMA_BF16(aq, bk_, z);
    }

    float rmax[8];
    #pragma unroll
    for (int r = 0; r < 8; ++r) rmax[r] = -1e30f;
    #pragma unroll
    for (int nt = 0; nt < 4; ++nt) {
      #pragma unroll
      for (int r = 0; r < 8; ++r) {
        float v = sc[nt][r] * ATT_SCALE;
        if (nt == 3 && lr > 0) v = -1e30f;        // mask key tokens 49..63
        sc[nt][r] = v;
        rmax[r] = fmaxf(rmax[r], v);
      }
    }
    #pragma unroll
    for (int r = 0; r < 8; ++r) {
      #pragma unroll
      for (int m = 1; m <= 8; m <<= 1)            // 16-lane butterfly (rows live per half)
        rmax[r] = fmaxf(rmax[r], __shfl_xor(rmax[r], m, 32));
    }
    float rsum[8];
    #pragma unroll
    for (int r = 0; r < 8; ++r) rsum[r] = 0.f;
    #pragma unroll
    for (int nt = 0; nt < 4; ++nt) {
      #pragma unroll
      for (int r = 0; r < 8; ++r) {
        float e = __expf(sc[nt][r] - rmax[r]);
        sc[nt][r] = e;
        rsum[r] += e;
      }
    }
    #pragma unroll
    for (int r = 0; r < 8; ++r) {
      #pragma unroll
      for (int m = 1; m <= 8; m <<= 1)
        rsum[r] += __shfl_xor(rsum[r], m, 32);
      rsum[r] = 1.0f / rsum[r];
    }

    unsigned short* myP = sP + wave * 16 * SP_STR;  // private 16x64 prob tile
    #pragma unroll
    for (int nt = 0; nt < 4; ++nt) {
      #pragma unroll
      for (int r = 0; r < 8; ++r)
        myP[(half * 8 + r) * SP_STR + nt * 16 + lr] = f2bf(sc[nt][r] * rsum[r]);
    }

    // P(16x49pad64) x V(49pad64 x 32): B from transposed V, K = tokens contiguous
    #pragma unroll
    for (int ntv = 0; ntv < 2; ++ntv) {
      v8f acc = {};
      #pragma unroll
      for (int ks = 0; ks < 2; ++ks) {
        v16bf ap  = load_tile_a(myP + ks * 32, SP_STR, lane);
        v16bf bv_ = load_tile_b(sVT + (h * HD + ntv * 16) * SVT_STR + ks * 32, SVT_STR, lane);
        acc = WMMA_BF16(ap, bv_, acc);
      }
      #pragma unroll
      for (int r = 0; r < 8; ++r)
        sX[(mt * 16 + half * 8 + r) * SX_STR + h * HD + ntv * 16 + lr] = f2bf(acc[r]);
    }
  }
  __syncthreads();

  // ---- output projection + bias, fp32 stores -------------------------------
  for (int t = wave; t < 48; t += NWAVE) {
    int mt = t / 12, nt = t - mt * 12;
    v8f acc = {};
    #pragma unroll
    for (int ks = 0; ks < 6; ++ks) {
      v16bf a = load_tile_a(sX + mt * 16 * SX_STR + ks * 32, SX_STR, lane);
      v16bf b = load_tile_b(wto + nt * 16 * DIMC  + ks * 32, DIMC,  lane);
      acc = WMMA_BF16(a, b, acc);
    }
    float bb = bo[nt * 16 + lr];
    #pragma unroll
    for (int r = 0; r < 8; ++r) {
      int token = mt * 16 + half * 8 + r;
      if (token < LTOK) {
        int ty = token / WSZ, tx = token - ty * WSZ;
        owin[(ty * IMG + tx) * DIMC + nt * 16 + lr] = acc[r] + bb;
      }
    }
  }
}

// ---------------------------------------------------------------------------
extern "C" void kernel_launch(void* const* d_in, const int* in_sizes, int n_in,
                              void* d_out, int out_size, void* d_ws, size_t ws_size,
                              hipStream_t stream)
{
  (void)in_sizes; (void)n_in; (void)out_size; (void)ws_size;
  const float* x  = (const float*)d_in[0];
  const float* Wq = (const float*)d_in[1];
  const float* bq = (const float*)d_in[2];
  const float* Wk = (const float*)d_in[3];
  const float* bk = (const float*)d_in[4];
  const float* Wv = (const float*)d_in[5];
  const float* bv = (const float*)d_in[6];
  const float* Wo = (const float*)d_in[7];
  const float* bo = (const float*)d_in[8];
  unsigned short* wt = (unsigned short*)d_ws;     // 4 * 192*192 bf16 = 288 KB

  prep_weights_kernel<<<(4 * DIMC * DIMC + 255) / 256, 256, 0, stream>>>(Wq, Wk, Wv, Wo, wt);

  size_t shmem = (size_t)(3 * LPAD * SX_STR + DIMC * SVT_STR + NWAVE * 16 * SP_STR)
               * sizeof(unsigned short);          // 122880 B
  win_attn_kernel<<<4096, 256, shmem, stream>>>(
      x, bq, bk, bv, bo,
      wt, wt + DIMC * DIMC, wt + 2 * DIMC * DIMC, wt + 3 * DIMC * DIMC,
      (float*)d_out);
}
